// TanhAttention_39573828666104
// MI455X (gfx1250) — matrix operand order repaired and checked by reading, not compiled, not run on hardware
//
#include <hip/hip_runtime.h>
#include <hip/hip_bf16.h>

// ---------------------------------------------------------------------------
// TanhAttention (additive attention), MI455X / gfx1250, wave32 + WMMA bf16.
//
//   item1 = x @ Wq + bq                       [B,Lq,H]
//   item2 = memory @ Wm                       [B,Lm,H]
//   scores[b,q,m] = sum_h wf[h]*tanh(item1[b,q,h]+item2[b,m,h])
//   attn = softmax(mask(scores))              [B,Lq,Lm]
//   out  = attn @ memory                      [B,Lq,D]
//
// Working set ~28MB -> fully L2 resident (192MB), so a multi-kernel pipeline
// through d_ws costs almost nothing in bandwidth. Pipeline:
//   P1: convert x/memory -> bf16 row-major         (A operands, b128 loads)
//   P2: transpose+convert memory -> memT bf16      (B operand of out GEMM)
//   P3: transpose+convert Wq/Wm  -> WqT/WmT bf16   (B operands of projections)
//   K1: projections via v_wmma_f32_16x16x32_bf16 (4 acc tiles / wave,
//       double-buffered fragment loads)
//   K2: scores + masked softmax; v_tanh_f32 trans pipe (268M tanh = the
//       true bottleneck; cannot be a matmul), attn emitted as bf16
//   K3: attn @ memory via v_wmma_f32_16x16x32_bf16 (same GEMM core)
// ---------------------------------------------------------------------------

typedef __attribute__((ext_vector_type(16))) __bf16 v16bf;
typedef __attribute__((ext_vector_type(8)))  __bf16 v8bf;
typedef __attribute__((ext_vector_type(8)))  float  v8f;

#define BATCH 8
#define LQ 512
#define LM 512
#define DIN 512
#define HID 128

__device__ __forceinline__ float fast_tanh(float x) {
#if __has_builtin(__builtin_amdgcn_tanhf)
    return __builtin_amdgcn_tanhf(x);
#else
    float t;
    // gfx1250 hardware tanh; v_nop satisfies the 1-op TRANS RAW hazard rule.
    asm volatile("v_tanh_f32 %0, %1\n\tv_nop" : "=v"(t) : "v"(x));
    return t;
#endif
}

// ---------------------------------------------------------------------------
// P1: f32 -> bf16 row-major copies of x and memory (8 elems / thread, b128 IO)
// ---------------------------------------------------------------------------
__global__ void __launch_bounds__(256)
cvt_rows_kernel(const float* __restrict__ x, const float* __restrict__ mem,
                __bf16* __restrict__ xb, __bf16* __restrict__ memb)
{
    const int id    = blockIdx.x * blockDim.x + threadIdx.x;   // 0..524287
    const int total = BATCH * LQ * DIN / 8;                    // 262144 per tensor
    const float4* __restrict__ src;
    __bf16* __restrict__ dst;
    int e;
    if (id < total) { src = (const float4*)x;   dst = xb;   e = id; }
    else            { src = (const float4*)mem; dst = memb; e = id - total; }
    const float4 v0 = src[2 * e + 0];
    const float4 v1 = src[2 * e + 1];
    v8bf o;
    o[0] = (__bf16)v0.x; o[1] = (__bf16)v0.y; o[2] = (__bf16)v0.z; o[3] = (__bf16)v0.w;
    o[4] = (__bf16)v1.x; o[5] = (__bf16)v1.y; o[6] = (__bf16)v1.z; o[7] = (__bf16)v1.w;
    *reinterpret_cast<v8bf*>(dst + 8 * e) = o;
}

// ---------------------------------------------------------------------------
// P2: memT[b][d][m] = (bf16) memory[b][m][d], LDS-tiled 32x32 transpose.
// ---------------------------------------------------------------------------
__global__ void __launch_bounds__(1024)
transpose_mem_kernel(const float* __restrict__ mem, __bf16* __restrict__ memT)
{
    __shared__ __bf16 tile[32][33];   // +1 pad: avoid LDS bank conflicts
    const int b  = blockIdx.z;
    const int m0 = blockIdx.x * 32;
    const int d0 = blockIdx.y * 32;
    const int tx = threadIdx.x, ty = threadIdx.y;
    tile[ty][tx] = (__bf16)mem[((size_t)(b * LM + m0 + ty)) * DIN + d0 + tx];
    __syncthreads();
    memT[((size_t)b * DIN + d0 + ty) * LM + m0 + tx] = tile[tx][ty];
}

// ---------------------------------------------------------------------------
// P3: WqT[h][k] = (bf16) Wq[k][h], WmT likewise (each 512x128 -> 128x512).
// ---------------------------------------------------------------------------
__global__ void __launch_bounds__(256)
transpose_w_kernel(const float* __restrict__ Wq, const float* __restrict__ Wm,
                   __bf16* __restrict__ WqT, __bf16* __restrict__ WmT)
{
    const int id  = blockIdx.x * blockDim.x + threadIdx.x;   // 0..131071
    const int per = DIN * HID;                               // 65536
    const float* __restrict__ W;
    __bf16* __restrict__ WT;
    int e;
    if (id < per) { W = Wq; WT = WqT; e = id; }
    else          { W = Wm; WT = WmT; e = id - per; }
    const int k = e / HID, h = e % HID;     // consecutive ids -> coalesced reads
    WT[h * DIN + k] = (__bf16)W[k * HID + h];
}

// ---------------------------------------------------------------------------
// Shared WMMA K-loop: C[4 tiles of 16x16] += A(16x512) x B(512x64).
//   A row-major bf16, stride 512. BT is B transposed (N-major), stride 512.
// Per K-step of 32: A fragment = 2x b128 loads (ISA 16-bit A 16x32 layout:
//   lane&15 = M row, half = lane>>4 selects K groups {half*8, 16+half*8});
// each B fragment = 2x b128 loads (lane&15 = N col, same K groups as A).
// Double-buffered: fragments for step k+1 are loaded while the 4 WMMAs of
// step k execute, so the WMMA pipe never waits on loadcnt==0.
// ---------------------------------------------------------------------------
__device__ __forceinline__ void
wmma_gemm4(const __bf16* __restrict__ Arow,   // A + rowA*512
           const __bf16* __restrict__ BTrow,  // BT + (nbase + (lane&15))*512
           const int half, v8f c[4])
{
    v16bf a[2];
    v16bf bb[2][4];

    auto load_frags = [&](int k0, int buf) {
        {
            const v8bf lo = *reinterpret_cast<const v8bf*>(Arow + k0 + half * 8);
            const v8bf hi = *reinterpret_cast<const v8bf*>(Arow + k0 + 16 + half * 8);
#pragma unroll
            for (int i = 0; i < 8; ++i) { a[buf][i] = lo[i]; a[buf][8 + i] = hi[i]; }
        }
#pragma unroll
        for (int t = 0; t < 4; ++t) {
            const __bf16* bp = BTrow + t * 16 * 512 + k0 + half * 8;
            const v8bf lo = *reinterpret_cast<const v8bf*>(bp);
            const v8bf hi = *reinterpret_cast<const v8bf*>(bp + 16);
#pragma unroll
            for (int i = 0; i < 8; ++i) { bb[buf][t][i] = lo[i]; bb[buf][t][8 + i] = hi[i]; }
        }
    };

    load_frags(0, 0);
#pragma unroll
    for (int it = 0; it < 16; ++it) {
        const int cur = it & 1;
        if (it < 15) {
            load_frags((it + 1) * 32, cur ^ 1);
            // Warm the near cache two steps ahead (WGP-scope prefetch).
            __builtin_prefetch(Arow + (it + 2) * 32, 0, 3);
        }
#pragma unroll
        for (int t = 0; t < 4; ++t)
            c[t] = __builtin_amdgcn_wmma_f32_16x16x32_bf16(
                       false, a[cur], false, bb[cur][t], (short)0, c[t], false, false);
    }
}

// ---------------------------------------------------------------------------
// K1: item1 = x@Wq + bq and item2 = memory@Wm. One wave = 16 rows x 64 cols.
// 1024 waves total (2 tensors x 256 row tiles x 2 col groups).
// ---------------------------------------------------------------------------
__global__ void __launch_bounds__(128)
proj_wmma_kernel(const __bf16* __restrict__ xb, const __bf16* __restrict__ memb,
                 const __bf16* __restrict__ WqT, const __bf16* __restrict__ WmT,
                 const float* __restrict__ bq,
                 float* __restrict__ item1, float* __restrict__ item2)
{
    const int wave = (blockIdx.x * blockDim.x + threadIdx.x) >> 5;  // 0..1023
    const int lane = threadIdx.x & 31;
    const int tensor = wave >> 9;          // 0 = item1, 1 = item2
    const int mt     = (wave >> 1) & 255;  // 256 row tiles of 16
    const int ng     = wave & 1;           // 2 col groups of 64 (H=128)

    const __bf16* __restrict__ A   = tensor ? memb : xb;
    const __bf16* __restrict__ BT  = tensor ? WmT  : WqT;
    float*        __restrict__ out = tensor ? item2 : item1;

    const int half = lane >> 4, l15 = lane & 15;
    const int rowA  = mt * 16 + l15;
    const int nbase = ng * 64;

    v8f c[4] = {};
    wmma_gemm4(A + (size_t)rowA * DIN, BT + (size_t)(nbase + l15) * DIN, half, c);

#pragma unroll
    for (int t = 0; t < 4; ++t) {
        const int col  = nbase + t * 16 + l15;
        const float bias = tensor ? 0.0f : bq[col];
#pragma unroll
        for (int p = 0; p < 8; ++p)
            out[(mt * 16 + p + half * 8) * HID + col] = c[t][p] + bias;
    }
}

// ---------------------------------------------------------------------------
// K2: one 256-thread block per (b,q) row: scores, mask, softmax, attn->bf16.
// Trans-pipe bound: 512*128 v_tanh_f32 per block (268M total).
// ---------------------------------------------------------------------------
__global__ void __launch_bounds__(256)
score_softmax_kernel(const float* __restrict__ item1,   // [B*LQ, H]
                     const float* __restrict__ item2,   // [B*LM, H]
                     const int*   __restrict__ mask,    // [B, LM]
                     const float* __restrict__ wf,      // [H]
                     __bf16*      __restrict__ attn)    // [B*LQ, LM]
{
    __shared__ float s_i1[HID];
    __shared__ float s_wf[HID];
    __shared__ float s_scores[LM];
    __shared__ float s_red[256];

    const int bq  = blockIdx.x;       // b*512 + q
    const int b   = bq >> 9;
    const int tid = threadIdx.x;

    if (tid < HID) {
        s_i1[tid] = item1[bq * HID + tid];
        s_wf[tid] = wf[tid];
    }
    __syncthreads();

#pragma unroll
    for (int rep = 0; rep < 2; ++rep) {
        const int m = tid + rep * 256;
        const float4* __restrict__ i2 =
            reinterpret_cast<const float4*>(&item2[(b * LM + m) * HID]);
        float s = 0.0f;
#pragma unroll 4
        for (int h4 = 0; h4 < HID / 4; ++h4) {
            const float4 v = i2[h4];
            const int h = 4 * h4;
            s = fmaf(s_wf[h + 0], fast_tanh(s_i1[h + 0] + v.x), s);
            s = fmaf(s_wf[h + 1], fast_tanh(s_i1[h + 1] + v.y), s);
            s = fmaf(s_wf[h + 2], fast_tanh(s_i1[h + 2] + v.z), s);
            s = fmaf(s_wf[h + 3], fast_tanh(s_i1[h + 3] + v.w), s);
        }
        if (mask[b * LM + m] == 0) s = -__builtin_inff();
        s_scores[m] = s;
    }
    __syncthreads();

    float mx = fmaxf(s_scores[tid], s_scores[tid + 256]);
    s_red[tid] = mx;
    __syncthreads();
    for (int off = 128; off > 0; off >>= 1) {
        if (tid < off) s_red[tid] = fmaxf(s_red[tid], s_red[tid + off]);
        __syncthreads();
    }
    mx = s_red[0];
    __syncthreads();

    const float e0 = __expf(s_scores[tid]       - mx);  // exp(-inf)=0 when masked
    const float e1 = __expf(s_scores[tid + 256] - mx);
    s_red[tid] = e0 + e1;
    __syncthreads();
    for (int off = 128; off > 0; off >>= 1) {
        if (tid < off) s_red[tid] += s_red[tid + off];
        __syncthreads();
    }
    const float inv = 1.0f / s_red[0];

    attn[(size_t)bq * LM + tid]       = (__bf16)(e0 * inv);
    attn[(size_t)bq * LM + tid + 256] = (__bf16)(e1 * inv);
}

// ---------------------------------------------------------------------------
// K3: out[b] = attn[b] @ memory[b], B from memT (N-major). One wave = 16x64.
// 2048 waves (8 batches x 32 row tiles x 8 col groups).
// ---------------------------------------------------------------------------
__global__ void __launch_bounds__(128)
out_gemm_kernel(const __bf16* __restrict__ attn,  // [B, LQ, LM] bf16
                const __bf16* __restrict__ memT,  // [B, D, LM]  bf16
                float*        __restrict__ out)   // [B, LQ, D]  f32
{
    const int wave = (blockIdx.x * blockDim.x + threadIdx.x) >> 5;  // 0..2047
    const int lane = threadIdx.x & 31;
    const int b  = wave >> 8;          // batch
    const int mt = (wave >> 3) & 31;   // q tile of 16
    const int ng = wave & 7;           // d group of 64

    const int half = lane >> 4, l15 = lane & 15;

    const __bf16* __restrict__ A  =
        attn + (size_t)b * LQ * LM + (size_t)(mt * 16 + l15) * LM;
    const __bf16* __restrict__ BT =
        memT + (size_t)b * DIN * LM + (size_t)(ng * 64 + l15) * LM;

    v8f c[4] = {};
    wmma_gemm4(A, BT, half, c);

    float* __restrict__ O = out + (size_t)b * LQ * DIN;
#pragma unroll
    for (int t = 0; t < 4; ++t) {
#pragma unroll
        for (int p = 0; p < 8; ++p)
            O[(mt * 16 + p + half * 8) * DIN + ng * 64 + t * 16 + l15] = c[t][p];
    }
}

// ---------------------------------------------------------------------------
// kernel_launch: d_in = x, memory, memory_mask, Wq, bq, Wm, wf
// d_ws layout (256B-aligned partitions, total ~20.3MB, L2-resident):
//   item1 f32 2MB | item2 f32 2MB | attn bf16 4MB | x bf16 4MB |
//   mem bf16 4MB | memT bf16 4MB | WqT bf16 128KB | WmT bf16 128KB
// ---------------------------------------------------------------------------
extern "C" void kernel_launch(void* const* d_in, const int* in_sizes, int n_in,
                              void* d_out, int out_size, void* d_ws, size_t ws_size,
                              hipStream_t stream) {
    const float* x      = (const float*)d_in[0];
    const float* memory = (const float*)d_in[1];
    const int*   mask   = (const int*)  d_in[2];
    const float* Wq     = (const float*)d_in[3];
    const float* bq     = (const float*)d_in[4];
    const float* Wm     = (const float*)d_in[5];
    const float* wf     = (const float*)d_in[6];
    float*       out    = (float*)d_out;

    char* ws = (char*)d_ws;
    const size_t ITEM  = (size_t)BATCH * LQ * HID * sizeof(float);    // 2MB
    const size_t ROWS  = (size_t)BATCH * LQ * DIN * sizeof(__bf16);   // 4MB
    const size_t WT_SZ = (size_t)DIN * HID * sizeof(__bf16);          // 128KB

    float*  item1 = (float*) (ws);
    float*  item2 = (float*) (ws + ITEM);
    __bf16* attn  = (__bf16*)(ws + 2 * ITEM);                 // 4MB
    __bf16* xb    = (__bf16*)(ws + 2 * ITEM + ROWS);
    __bf16* memb  = (__bf16*)(ws + 2 * ITEM + 2 * ROWS);
    __bf16* memT  = (__bf16*)(ws + 2 * ITEM + 3 * ROWS);
    __bf16* WqT   = (__bf16*)(ws + 2 * ITEM + 4 * ROWS);
    __bf16* WmT   = (__bf16*)(ws + 2 * ITEM + 4 * ROWS + WT_SZ);

    // P1: 2 x 4096x512 f32 -> bf16 (8 elems/thread)
    cvt_rows_kernel<<<2048, 256, 0, stream>>>(x, memory, xb, memb);
    // P2: memory -> memT (bf16, transposed per batch)
    transpose_mem_kernel<<<dim3(LM / 32, DIN / 32, BATCH), dim3(32, 32), 0, stream>>>(memory, memT);
    // P3: Wq/Wm -> WqT/WmT (bf16, transposed)
    transpose_w_kernel<<<512, 256, 0, stream>>>(Wq, Wm, WqT, WmT);

    // K1: 1024 waves, 4 waves/block
    proj_wmma_kernel<<<256, 128, 0, stream>>>(xb, memb, WqT, WmT, bq, item1, item2);
    // K2: one block per (b,q) row
    score_softmax_kernel<<<BATCH * LQ, 256, 0, stream>>>(item1, item2, mask, wf, attn);
    // K3: 2048 waves, 4 waves/block
    out_gemm_kernel<<<512, 128, 0, stream>>>(attn, memT, out);
}
